// FaceClassificationGNN_10058813407637
// MI455X (gfx1250) — matrix operand order repaired and checked by reading, not compiled, not run on hardware
//
#include <hip/hip_runtime.h>
#include <hip/hip_bf16.h>
#include <stdint.h>

#define NN 50000
#define NE 500000
#define NE2 (NE + NN)
#define NEGS 0.2f
#define BNEPS 1e-5f

typedef __attribute__((ext_vector_type(16))) __bf16 v16bf;
typedef __attribute__((ext_vector_type(8)))  __bf16 v8bf;
typedef __attribute__((ext_vector_type(8)))  float  v8f;

// ---------- helpers ----------
__device__ __forceinline__ __bf16 to_bf16(float f) {
  unsigned u = __float_as_uint(f);
  unsigned r = u + 0x7FFFu + ((u >> 16) & 1u);   // round-to-nearest-even
  unsigned short hs = (unsigned short)(r >> 16);
  __bf16 b; __builtin_memcpy(&b, &hs, 2); return b;
}
// monotone float<->uint mapping for atomic max on floats (any sign)
__device__ __forceinline__ unsigned fenc(float f) {
  unsigned u = __float_as_uint(f);
  return (u & 0x80000000u) ? ~u : (u | 0x80000000u);
}
__device__ __forceinline__ float fdec(unsigned u) {
  unsigned v = (u & 0x80000000u) ? (u & 0x7FFFFFFFu) : ~u;
  return __uint_as_float(v);
}

// ---------- trivial utility kernels ----------
__global__ void k_zero_f32(float* p, long n) {
  long t = (long)blockIdx.x * blockDim.x + threadIdx.x;
  if (t < n) p[t] = 0.0f;
}
__global__ void k_zero_u32(unsigned* p, long n) {
  long t = (long)blockIdx.x * blockDim.x + threadIdx.x;
  if (t < n) p[t] = 0u;
}
__global__ void k_cvt_bf16(const float* __restrict__ s, __bf16* __restrict__ d, long n) {
  long t = (long)blockIdx.x * blockDim.x + threadIdx.x;
  if (t < n) d[t] = to_bf16(s[t]);
}

// h = relu(x @ W_in + b_in)   [N,12]x[12,64]
__global__ void k_lin_in(const float* __restrict__ x, const float* __restrict__ W,
                         const float* __restrict__ b, float* __restrict__ h) {
  long t = (long)blockIdx.x * blockDim.x + threadIdx.x;
  if (t >= (long)NN * 64) return;
  int n = (int)(t >> 6), c = (int)(t & 63);
  float acc = b[c];
  const float* xr = x + (long)n * 12;
  #pragma unroll
  for (int k = 0; k < 12; ++k) acc += xr[k] * W[k * 64 + c];
  h[t] = acc > 0.0f ? acc : 0.0f;
}

// ---------- self-loop edge_attr = mean of incoming ----------
__global__ void k_loop_ea(const float* __restrict__ ea, const int* __restrict__ dst,
                          float* __restrict__ cnt, float* __restrict__ lsum) {
  int e = blockIdx.x * blockDim.x + threadIdx.x;
  if (e >= NE) return;
  int d = dst[e];
  atomicAdd(&cnt[d], 1.0f);
  atomicAdd(&lsum[(long)d * 2 + 0], ea[(long)e * 2 + 0]);
  atomicAdd(&lsum[(long)d * 2 + 1], ea[(long)e * 2 + 1]);
}
__global__ void k_build_ea(const float* __restrict__ ea, const float* __restrict__ cnt,
                           const float* __restrict__ lsum, float* __restrict__ eaa) {
  int e = blockIdx.x * blockDim.x + threadIdx.x;
  if (e >= NE2) return;
  if (e < NE) {
    eaa[(long)e * 2 + 0] = ea[(long)e * 2 + 0];
    eaa[(long)e * 2 + 1] = ea[(long)e * 2 + 1];
  } else {
    int i = e - NE;
    float c = cnt[i]; float dn = c > 1.0f ? c : 1.0f;
    eaa[(long)e * 2 + 0] = lsum[(long)i * 2 + 0] / dn;
    eaa[(long)e * 2 + 1] = lsum[(long)i * 2 + 1] / dn;
  }
}

// ---------- weight pack: fp32 [K,M] -> bf16 in WMMA B-fragment order ----------
// packed index t = ((kb*ctiles)+ct)*512 + lane*16 + j ; k = kb*32+(lane>>4)*16+j ; n = ct*16+(lane&15)
__global__ void k_pack_w(const float* __restrict__ W, __bf16* __restrict__ Bp, int K, int M) {
  int t = blockIdx.x * blockDim.x + threadIdx.x;
  int total = K * M;
  if (t >= total) return;
  int ctiles = M >> 4;
  int kb  = t / (ctiles * 512);
  int rem = t - kb * ctiles * 512;
  int ct  = rem >> 9;
  int q   = rem & 511;
  int lane = q >> 4, j = q & 15;
  int k = kb * 32 + (lane >> 4) * 16 + j;
  int n = ct * 16 + (lane & 15);
  Bp[t] = to_bf16(W[(long)k * M + n]);
}

// ---------- BF16 WMMA GEMM:  C[rows,M] = A[rows,K](bf16) x Bpacked + bias ----------
// Each wave owns 1 row tile x CT column tiles. Per K-step: load A once + all CT
// B fragments up front (one clause, partial waits), then CT back-to-back WMMAs
// reusing the A registers -> load latency hidden behind matrix math.
template <int CT>
__global__ void k_gemm_bf16(const __bf16* __restrict__ A, const __bf16* __restrict__ Bp,
                            const float* __restrict__ bias, float* __restrict__ C,
                            int rowTiles, int K, int M, int relu) {
  int lane = threadIdx.x & 31;
  int wave = threadIdx.x >> 5;
  int rt  = blockIdx.y * 8 + wave;
  int ctg = blockIdx.x;                          // group of CT consecutive col tiles
  if (rt >= rowTiles) return;                    // uniform per wave -> EXEC stays all-1s
  int ctiles = M >> 4;
  int m  = lane & 15;
  int kg = lane >> 4;                            // 0 or 1
  const __bf16* arow  = A + (long)(rt * 16 + m) * K + kg * 16;
  const __bf16* bbase = Bp + (long)ctg * CT * 512 + (long)lane * 16;
  v8f acc[CT];
  #pragma unroll
  for (int c = 0; c < CT; ++c) acc[c] = (v8f){};
  int kblocks = K >> 5;
  for (int kb = 0; kb < kblocks; ++kb) {
    if (kb + 1 < kblocks)
      __builtin_prefetch(bbase + (long)(kb + 1) * ctiles * 512, 0, 1);
    // --- issue ALL loads for this K-step first ---
    v8bf alo = *(const v8bf*)(arow + kb * 32);        // K = base..base+7
    v8bf ahi = *(const v8bf*)(arow + kb * 32 + 8);    // K = base+8..base+15
    const __bf16* bk = bbase + (long)kb * ctiles * 512;
    v16bf bfr[CT];
    #pragma unroll
    for (int c = 0; c < CT; ++c)
      bfr[c] = *(const v16bf*)(bk + (long)c * 512);   // consecutive 1KB blocks
    v16bf a;
    #pragma unroll
    for (int j = 0; j < 8; ++j) { a[j] = alo[j]; a[j + 8] = ahi[j]; }
    // --- then the WMMA chain (A registers reused CT times) ---
    #pragma unroll
    for (int c = 0; c < CT; ++c)
      acc[c] = __builtin_amdgcn_wmma_f32_16x16x32_bf16(false, a, false, bfr[c],
                                                       (short)0, acc[c], false, false);
  }
  #pragma unroll
  for (int c = 0; c < CT; ++c) {
    int col = (ctg * CT + c) * 16 + (lane & 15);
    float bi = bias[col];
    #pragma unroll
    for (int r = 0; r < 8; ++r) {
      int row = rt * 16 + kg * 8 + r;
      float v = acc[c][r] + bi;
      if (relu) v = v > 0.0f ? v : 0.0f;
      C[(long)row * M + col] = v;
    }
  }
}

// ---------- GATv2 edge passes ----------
__device__ __forceinline__ void edge_sd(const int* __restrict__ s, const int* __restrict__ d,
                                        int e, int& se, int& de) {
  if (e < NE) { se = s[e]; de = d[e]; } else { se = de = e - NE; }
}

// pass1: alpha[e,h] = sum_c leaky(xl[s]+xr[d]+ea@We) * att ; atomic segment-max
__global__ void k_edge_alpha(const float* __restrict__ xl, const float* __restrict__ xr,
                             const float* __restrict__ eaa, const float* __restrict__ We,
                             const float* __restrict__ att, const int* __restrict__ srcI,
                             const int* __restrict__ dstI, float* __restrict__ alp,
                             unsigned* __restrict__ amax, int H) {
  long t = (long)blockIdx.x * blockDim.x + threadIdx.x;
  if (t >= (long)NE2 * H) return;
  int e = (int)(t / H), h = (int)(t % H);
  int s, d; edge_sd(srcI, dstI, e, s, d);
  int HC = H * 64;
  const float* pl = xl + (long)s * HC + h * 64;
  const float* pr = xr + (long)d * HC + h * 64;
  const float* w0 = We + h * 64;
  const float* w1 = We + HC + h * 64;
  const float* at = att + h * 64;
  float e0 = eaa[(long)e * 2], e1 = eaa[(long)e * 2 + 1];
  float acc = 0.0f;
  #pragma unroll 4
  for (int c = 0; c < 64; ++c) {
    float mv = pl[c] + pr[c] + e0 * w0[c] + e1 * w1[c];
    mv = mv > 0.0f ? mv : NEGS * mv;
    acc += mv * at[c];
  }
  alp[t] = acc;
  atomicMax(&amax[(long)d * H + h], fenc(acc));
}

// pass2: ex = exp(alpha - amax[dst]); den[dst] += ex   (ex overwrites alpha)
__global__ void k_edge_exp(const int* __restrict__ srcI, const int* __restrict__ dstI,
                           float* __restrict__ alp, const unsigned* __restrict__ amax,
                           float* __restrict__ den, int H) {
  long t = (long)blockIdx.x * blockDim.x + threadIdx.x;
  if (t >= (long)NE2 * H) return;
  int e = (int)(t / H), h = (int)(t % H);
  int s, d; edge_sd(srcI, dstI, e, s, d);
  float ex = __expf(alp[t] - fdec(amax[(long)d * H + h]));
  alp[t] = ex;
  atomicAdd(&den[(long)d * H + h], ex);
}

// pass3: out[dst] += xl[src] * ex/(den+1e-16)
__global__ void k_edge_scatter(const float* __restrict__ xl, const float* __restrict__ ex,
                               const float* __restrict__ den, const int* __restrict__ srcI,
                               const int* __restrict__ dstI, float* __restrict__ out, int H) {
  long t = (long)blockIdx.x * blockDim.x + threadIdx.x;
  int HC = H * 64;
  if (t >= (long)NE2 * HC) return;
  int e = (int)(t / HC);
  int r = (int)(t % HC);
  int h = r >> 6;
  int s, d; edge_sd(srcI, dstI, e, s, d);
  float a = ex[(long)e * H + h] / (den[(long)d * H + h] + 1e-16f);
  atomicAdd(&out[(long)d * HC + r], xl[(long)s * HC + r] * a);
}

// ---------- BatchNorm (training-style, per-column stats over N) ----------
__global__ void k_bn_stats(const float* __restrict__ x, float* __restrict__ mu,
                           float* __restrict__ var, int rows, int M) {
  __shared__ float s1[256], s2[256];
  int col = blockIdx.x;
  float a = 0.0f, b = 0.0f;
  for (int r = threadIdx.x; r < rows; r += 256) {
    float v = x[(long)r * M + col];
    a += v; b += v * v;
  }
  s1[threadIdx.x] = a; s2[threadIdx.x] = b;
  __syncthreads();
  for (int s = 128; s > 0; s >>= 1) {
    if (threadIdx.x < s) { s1[threadIdx.x] += s1[threadIdx.x + s]; s2[threadIdx.x] += s2[threadIdx.x + s]; }
    __syncthreads();
  }
  if (threadIdx.x == 0) {
    float m = s1[0] / rows;
    mu[col] = m;
    var[col] = s2[0] / rows - m * m;
  }
}
__global__ void k_bn_apply(float* __restrict__ x, const float* __restrict__ mu,
                           const float* __restrict__ var, const float* __restrict__ gamma,
                           const float* __restrict__ beta, long total, int M, int relu) {
  long t = (long)blockIdx.x * blockDim.x + threadIdx.x;
  if (t >= total) return;
  int col = (int)(t % M);
  float v = gamma[col] * (x[t] - mu[col]) * __frsqrt_rn(var[col] + BNEPS) + beta[col];
  if (relu) v = v > 0.0f ? v : 0.0f;
  x[t] = v;
}

// ---------- classifier final 32 -> 2 ----------
__global__ void k_cls_out(const float* __restrict__ hid, const float* __restrict__ W,
                          const float* __restrict__ b, float* __restrict__ out) {
  long t = (long)blockIdx.x * blockDim.x + threadIdx.x;
  if (t >= (long)NN * 2) return;
  int n = (int)(t >> 1), o = (int)(t & 1);
  float acc = b[o];
  const float* hr = hid + (long)n * 32;
  #pragma unroll
  for (int c = 0; c < 32; ++c) acc += hr[c] * W[c * 2 + o];
  out[t] = acc;
}

extern "C" void kernel_launch(void* const* d_in, const int* in_sizes, int n_in,
                              void* d_out, int out_size, void* d_ws, size_t ws_size,
                              hipStream_t stream) {
  (void)in_sizes; (void)n_in; (void)out_size; (void)ws_size;
  const float* x    = (const float*)d_in[0];
  const float* ea   = (const float*)d_in[1];
  const float* W_in = (const float*)d_in[2];
  const float* b_in = (const float*)d_in[3];
  const float *Wl[3], *bl[3], *Wr[3], *br[3], *We[3], *att[3], *gm[3], *bt[3];
  for (int l = 0; l < 3; ++l) {
    int b4 = 4 + 9 * l;
    Wl[l]  = (const float*)d_in[b4 + 0]; bl[l] = (const float*)d_in[b4 + 1];
    Wr[l]  = (const float*)d_in[b4 + 2]; br[l] = (const float*)d_in[b4 + 3];
    We[l]  = (const float*)d_in[b4 + 4]; att[l] = (const float*)d_in[b4 + 5];
    // bg (b4+6) cancels inside BatchNorm (mean-shift invariance) -> skipped
    gm[l]  = (const float*)d_in[b4 + 7]; bt[l] = (const float*)d_in[b4 + 8];
  }
  const float* Wc1 = (const float*)d_in[31];
  const float* bc1 = (const float*)d_in[32];
  const float* Wc2 = (const float*)d_in[33];
  const float* bc2 = (const float*)d_in[34];
  const int*   ei  = (const int*)d_in[35];
  const int* srcI = ei;
  const int* dstI = ei + NE;
  float* out = (float*)d_out;

  // --- workspace carve (all 256B aligned) ---
  char* base = (char*)d_ws; size_t off = 0;
  auto alloc = [&](size_t bytes) -> char* {
    char* p = base + off; off += (bytes + 255) & ~(size_t)255; return p;
  };
  float*    h    = (float*)alloc((size_t)NN * 256 * 4);
  float*    xl   = (float*)alloc((size_t)NN * 256 * 4);
  float*    xr   = (float*)alloc((size_t)NN * 256 * 4);
  __bf16*   hb   = (__bf16*)alloc((size_t)NN * 256 * 2);
  float*    alp  = (float*)alloc((size_t)NE2 * 4 * 4);
  unsigned* amax = (unsigned*)alloc((size_t)NN * 4 * 4);
  float*    den  = (float*)alloc((size_t)NN * 4 * 4);
  float*    eaa  = (float*)alloc((size_t)NE2 * 2 * 4);
  float*    cnt  = (float*)alloc((size_t)NN * 4);
  float*    lsum = (float*)alloc((size_t)NN * 2 * 4);
  __bf16*   wpL  = (__bf16*)alloc((size_t)256 * 256 * 2);
  __bf16*   wpR  = (__bf16*)alloc((size_t)256 * 256 * 2);
  float*    mu   = (float*)alloc(256 * 4);
  float*    var  = (float*)alloc(256 * 4);

  const int TB = 256;
  auto blk = [](long n) { return (unsigned)((n + 255) / 256); };

  // self-loop edge attributes (fill_value='mean')
  k_zero_f32<<<blk(NN), TB, 0, stream>>>(cnt, NN);
  k_zero_f32<<<blk((long)NN * 2), TB, 0, stream>>>(lsum, (long)NN * 2);
  k_loop_ea<<<blk(NE), TB, 0, stream>>>(ea, dstI, cnt, lsum);
  k_build_ea<<<blk(NE2), TB, 0, stream>>>(ea, cnt, lsum, eaa);

  // input projection
  k_lin_in<<<blk((long)NN * 64), TB, 0, stream>>>(x, W_in, b_in, h);

  const int inD[3] = {64, 256, 256};
  const int Hh[3]  = {4, 4, 1};
  for (int l = 0; l < 3; ++l) {
    int K = inD[l], H = Hh[l], M = H * 64;
    k_cvt_bf16<<<blk((long)NN * K), TB, 0, stream>>>(h, hb, (long)NN * K);
    k_pack_w<<<blk((long)K * M), TB, 0, stream>>>(Wl[l], wpL, K, M);
    k_pack_w<<<blk((long)K * M), TB, 0, stream>>>(Wr[l], wpR, K, M);
    dim3 g(M / 64, (NN / 16 + 7) / 8);           // CT=4 col tiles per wave
    k_gemm_bf16<4><<<g, TB, 0, stream>>>(hb, wpL, bl[l], xl, NN / 16, K, M, 0);
    k_gemm_bf16<4><<<g, TB, 0, stream>>>(hb, wpR, br[l], xr, NN / 16, K, M, 0);
    // h is free now -> becomes the scatter target
    k_zero_f32<<<blk((long)NN * M), TB, 0, stream>>>(h, (long)NN * M);
    k_zero_u32<<<blk((long)NN * H), TB, 0, stream>>>(amax, (long)NN * H);
    k_zero_f32<<<blk((long)NN * H), TB, 0, stream>>>(den, (long)NN * H);
    k_edge_alpha<<<blk((long)NE2 * H), TB, 0, stream>>>(xl, xr, eaa, We[l], att[l],
                                                        srcI, dstI, alp, amax, H);
    k_edge_exp<<<blk((long)NE2 * H), TB, 0, stream>>>(srcI, dstI, alp, amax, den, H);
    k_edge_scatter<<<blk((long)NE2 * M), TB, 0, stream>>>(xl, alp, den, srcI, dstI, h, H);
    // BatchNorm (+ReLU for layers 0,1), in place on h
    k_bn_stats<<<M, TB, 0, stream>>>(h, mu, var, NN, M);
    k_bn_apply<<<blk((long)NN * M), TB, 0, stream>>>(h, mu, var, gm[l], bt[l],
                                                     (long)NN * M, M, l < 2 ? 1 : 0);
  }

  // classifier: relu(h @ Wc1 + bc1) @ Wc2 + bc2
  k_cvt_bf16<<<blk((long)NN * 64), TB, 0, stream>>>(h, hb, (long)NN * 64);
  k_pack_w<<<blk((long)64 * 32), TB, 0, stream>>>(Wc1, wpL, 64, 32);
  dim3 g1(1, (NN / 16 + 7) / 8);                 // M=32 -> CT=2 col tiles per wave
  k_gemm_bf16<2><<<g1, TB, 0, stream>>>(hb, wpL, bc1, xl, NN / 16, 64, 32, 1);
  k_cls_out<<<blk((long)NN * 2), TB, 0, stream>>>(xl, Wc2, bc2, out);
}
